// GCN_MLC_78116865179887
// MI455X (gfx1250) — matrix operand order repaired and checked by reading, not compiled, not run on hardware
//
#include <hip/hip_runtime.h>
#include <hip/hip_bf16.h>

// ---------------------------------------------------------------------------
// GCN (2-layer) for MI455X / gfx1250.
//   - GEMMs via v_wmma_f32_16x16x32_bf16 with fragment-major pre-packed
//     operands (1 v16bf load per matrix operand per WMMA, zero shuffles)
//   - edge aggregation via wave-per-edge coalesced gather + HW f32 atomics
// ---------------------------------------------------------------------------

#define N_NODES   100000
#define N_EDGES   3200000
#define N_FEAT    128
#define HIDDEN    64
#define N_CLASSES 32

typedef __attribute__((ext_vector_type(16))) __bf16 v16bf;
typedef __attribute__((ext_vector_type(8)))  float  v8f;

__device__ __forceinline__ void atomic_add_f32(float* p, float v) {
    // lowers to global_atomic_add_f32 (no CAS loop)
    unsafeAtomicAdd(p, v);
}

// ---------------- degree / norm precompute ----------------

__global__ void k_init_deg(float* __restrict__ deg, int n) {
    int i = blockIdx.x * blockDim.x + threadIdx.x;
    if (i < n) deg[i] = 1.0f;   // self-loop weight
}

__global__ void k_deg_edges(const float* __restrict__ A0, const float* __restrict__ A1,
                            const float* __restrict__ wsv,
                            const long long* __restrict__ cols,
                            float* __restrict__ deg, int ne) {
    int e = blockIdx.x * blockDim.x + threadIdx.x;
    if (e >= ne) return;
    float w = wsv[0] * A0[e] + wsv[1] * A1[e];
    atomic_add_f32(deg + cols[e], w);
}

__global__ void k_dinv(const float* __restrict__ deg, float* __restrict__ dinv, int n) {
    int i = blockIdx.x * blockDim.x + threadIdx.x;
    if (i >= n) return;
    float d = deg[i];
    dinv[i] = (d > 0.0f) ? rsqrtf(d) : 0.0f;
}

__global__ void k_norm(const float* __restrict__ A0, const float* __restrict__ A1,
                       const float* __restrict__ wsv,
                       const long long* __restrict__ rows,
                       const long long* __restrict__ cols,
                       const float* __restrict__ dinv,
                       float* __restrict__ norm, int ne) {
    int e = blockIdx.x * blockDim.x + threadIdx.x;
    if (e >= ne) return;
    float w = wsv[0] * A0[e] + wsv[1] * A1[e];
    norm[e] = dinv[rows[e]] * w * dinv[cols[e]];
}

// ---------------- fragment-major packing -----------------------------------
// A (activation) fragments, layout [m_tiles][K/32][lane 0..31][16 bf16]:
//   lane l (half = l>>4), element j:
//     j<8 : K = kc*32 +      half*8 + j      (WMMA 16-bit A VGPR0..3 region)
//     j>=8: K = kc*32 + 16 + half*8 + (j-8)  (VGPR4..7 region)
//   M = tm*16 + (l&15)
// B (weight) fragments, layout [n_tiles][K/32][lane][16 bf16]:
//   K = kc*32 + (l>>4)*8 + (j>>3)*16 + (j&7),  N = tn*16 + (l&15)

template <int K>
__global__ void k_pack_act(const float* __restrict__ src, __bf16* __restrict__ dst,
                           int total) {
    int p = blockIdx.x * blockDim.x + threadIdx.x;
    if (p >= total) return;
    constexpr int KC = K / 32;
    const int j    = p & 15;
    const int l    = (p >> 4) & 31;
    const int tile = p >> 9;                 // tm*KC + kc
    const int tm   = tile / KC;
    const int kc   = tile - tm * KC;
    const int m    = tm * 16 + (l & 15);
    const int k    = kc * 32 + ((l >> 4) * 8) + ((j >> 3) * 16) + (j & 7);
    dst[p] = (__bf16)src[(size_t)m * K + k];
}

// fused: bias + relu + bf16 + fragment pack (layer-2 activations, K = HIDDEN)
__global__ void k_pack_relu(const float* __restrict__ agg, const float* __restrict__ b1,
                            __bf16* __restrict__ dst, int total) {
    int p = blockIdx.x * blockDim.x + threadIdx.x;
    if (p >= total) return;
    constexpr int K = HIDDEN, KC = K / 32;
    const int j    = p & 15;
    const int l    = (p >> 4) & 31;
    const int tile = p >> 9;
    const int tm   = tile / KC;
    const int kc   = tile - tm * KC;
    const int m    = tm * 16 + (l & 15);
    const int k    = kc * 32 + ((l >> 4) * 8) + ((j >> 3) * 16) + (j & 7);
    float v = agg[(size_t)m * K + k] + b1[k];
    dst[p] = (__bf16)(v > 0.0f ? v : 0.0f);
}

template <int K, int NC>
__global__ void k_pack_w(const float* __restrict__ W, __bf16* __restrict__ dst,
                         int total) {
    int p = blockIdx.x * blockDim.x + threadIdx.x;
    if (p >= total) return;
    constexpr int KC = K / 32;
    const int j    = p & 15;
    const int l    = (p >> 4) & 31;
    const int tile = p >> 9;                 // tn*KC + kc
    const int tn   = tile / KC;
    const int kc   = tile - tn * KC;
    const int n    = tn * 16 + (l & 15);
    const int k    = kc * 32 + ((l >> 4) * 8) + ((j >> 3) * 16) + (j & 7);
    dst[p] = (__bf16)W[(size_t)k * NC + n];
}

// ---------------- WMMA GEMM on packed fragments -----------------------------
// One wave per 16-row strip: A fragment loaded once per k-chunk and reused
// across all NC/16 n-tiles. Inner loop = 1 v16bf load + 1 v_wmma.

template <int K, int NC>
__global__ void k_gemm_packed(const __bf16* __restrict__ Ap, const __bf16* __restrict__ Bp,
                              float* __restrict__ C, int m_tiles) {
    constexpr int KC = K / 32, NT = NC / 16;
    const int lane = threadIdx.x & 31;
    const int tm   = (blockIdx.x * blockDim.x + threadIdx.x) >> 5;
    if (tm >= m_tiles) return;               // wave-uniform exit (EXEC all-1 for WMMA)

    const v16bf* ap = (const v16bf*)Ap + (size_t)tm * KC * 32 + lane;
    const v16bf* bp = (const v16bf*)Bp + lane;

    v8f acc[NT];
#pragma unroll
    for (int tn = 0; tn < NT; ++tn) acc[tn] = (v8f){};

#pragma unroll
    for (int kc = 0; kc < KC; ++kc) {
        v16bf a = ap[(size_t)kc * 32];
#pragma unroll
        for (int tn = 0; tn < NT; ++tn) {
            v16bf b = bp[(size_t)(tn * KC + kc) * 32];
            acc[tn] = __builtin_amdgcn_wmma_f32_16x16x32_bf16(
                          false, a, false, b, (short)0, acc[tn], false, false);
        }
    }

    // C/D layout: VGPR r -> M = r (lanes 0-15), M = r+8 (lanes 16-31)
    const int nlo  = lane & 15;
    const int mrow = tm * 16 + (lane >> 4) * 8;
#pragma unroll
    for (int tn = 0; tn < NT; ++tn)
#pragma unroll
        for (int r = 0; r < 8; ++r)
            C[(size_t)(mrow + r) * NC + tn * 16 + nlo] = acc[tn][r];
}

// ---------------- self-loop init / output init ----------------

// agg[i,f] = dinv[i]^2 * h[i,f]   (self-loop term; also zero-initializes agg)
__global__ void k_self_init(const float* __restrict__ h, const float* __restrict__ dinv,
                            float* __restrict__ agg, int total, int nf) {
    int i = blockIdx.x * blockDim.x + threadIdx.x;
    if (i >= total) return;
    int node = i / nf;
    float d = dinv[node];
    agg[i] = d * d * h[i];
}

// out[i,c] = dinv[i]^2 * g2[i,c] + b2[c]
__global__ void k_out_init(const float* __restrict__ g2, const float* __restrict__ dinv,
                           const float* __restrict__ b2, float* __restrict__ out,
                           int total, int nc) {
    int i = blockIdx.x * blockDim.x + threadIdx.x;
    if (i >= total) return;
    int node = i / nc;
    float d = dinv[node];
    out[i] = d * d * g2[i] + b2[i % nc];
}

// ---------------- edge aggregation: out[col] += norm * src[row] -------------
// One wave per edge: coalesced NF-wide gather of src[row], HW f32 atomics
// into out[col] (targets are L2-resident: N*NF*4 <= 25.6 MB << 192 MB L2).

template <int NF>
__global__ void k_agg_edges(const float* __restrict__ src,
                            const long long* __restrict__ rows,
                            const long long* __restrict__ cols,
                            const float* __restrict__ norm,
                            float* __restrict__ out, int ne) {
    int wid  = (blockIdx.x * blockDim.x + threadIdx.x) >> 5;
    int lane = threadIdx.x & 31;
    if (wid >= ne) return;
    const long long s = rows[wid];
    const long long t = cols[wid];
    const float     w = norm[wid];
    const float* sp = src + (size_t)s * NF;
    float*       op = out + (size_t)t * NF;
#pragma unroll
    for (int f = lane; f < NF; f += 32)
        atomic_add_f32(op + f, w * sp[f]);
}

// ---------------------------------------------------------------------------

extern "C" void kernel_launch(void* const* d_in, const int* in_sizes, int n_in,
                              void* d_out, int out_size, void* d_ws, size_t ws_size,
                              hipStream_t stream) {
    (void)in_sizes; (void)n_in; (void)out_size; (void)ws_size;

    const float*     x   = (const float*)d_in[0];
    const long long* ei  = (const long long*)d_in[1];   // int64 [2, E]
    const float*     A0  = (const float*)d_in[2];
    const float*     A1  = (const float*)d_in[3];
    const float*     wsv = (const float*)d_in[4];
    const float*     W1  = (const float*)d_in[5];
    const float*     b1  = (const float*)d_in[6];
    const float*     W2  = (const float*)d_in[7];
    const float*     b2  = (const float*)d_in[8];
    float*           out = (float*)d_out;

    const long long* rows = ei;             // edge_index[0] (source)
    const long long* cols = ei + N_EDGES;   // edge_index[1] (target)

    // ---- workspace carve-out (256B aligned); aliased where lifetimes allow
    char*  base = (char*)d_ws;
    size_t off  = 0;
    auto carve = [&](size_t bytes) {
        char* p = base + off;
        off += (bytes + 255) & ~(size_t)255;
        return p;
    };
    float*  deg  = (float*)carve(N_NODES * sizeof(float));
    float*  dinv = (float*)carve(N_NODES * sizeof(float));
    float*  norm = (float*)carve((size_t)N_EDGES * sizeof(float));
    // region A: xb packed (bf16 [N,128]) later reused as g2 (f32 [N,32])
    char*   regA = carve((size_t)N_NODES * N_FEAT * sizeof(__bf16));
    // region B: h1 (f32 [N,64]) later reused as h2b packed (bf16 [N,64])
    char*   regB = carve((size_t)N_NODES * HIDDEN * sizeof(float));
    float*  agg1 = (float*)carve((size_t)N_NODES * HIDDEN * sizeof(float));
    __bf16* Wb1  = (__bf16*)carve((size_t)N_FEAT * HIDDEN * sizeof(__bf16));
    __bf16* Wb2  = (__bf16*)carve((size_t)HIDDEN * N_CLASSES * sizeof(__bf16));

    __bf16* xb  = (__bf16*)regA;
    float*  g2  = (float*)regA;     // alive only after xb is dead
    float*  h1  = (float*)regB;
    __bf16* h2b = (__bf16*)regB;    // alive only after h1 is dead

    const int TPB = 256;
    auto blk = [](long long n, int t) { return (unsigned)((n + t - 1) / t); };

    // ---- 1) degrees + symmetric norm
    k_init_deg <<<blk(N_NODES, TPB), TPB, 0, stream>>>(deg, N_NODES);
    k_deg_edges<<<blk(N_EDGES, TPB), TPB, 0, stream>>>(A0, A1, wsv, cols, deg, N_EDGES);
    k_dinv     <<<blk(N_NODES, TPB), TPB, 0, stream>>>(deg, dinv, N_NODES);
    k_norm     <<<blk(N_EDGES, TPB), TPB, 0, stream>>>(A0, A1, wsv, rows, cols, dinv, norm, N_EDGES);

    // ---- 2) fragment-major bf16 packing of x, W1, W2
    k_pack_act<N_FEAT><<<blk((long long)N_NODES * N_FEAT, TPB), TPB, 0, stream>>>(
        x, xb, N_NODES * N_FEAT);
    k_pack_w<N_FEAT, HIDDEN><<<blk(N_FEAT * HIDDEN, TPB), TPB, 0, stream>>>(
        W1, Wb1, N_FEAT * HIDDEN);
    k_pack_w<HIDDEN, N_CLASSES><<<blk(HIDDEN * N_CLASSES, TPB), TPB, 0, stream>>>(
        W2, Wb2, HIDDEN * N_CLASSES);

    // ---- 3) layer 1: h1 = x @ W1  (WMMA, wave per 16-row strip)
    const int m_tiles = N_NODES / 16;                 // 6250 (exact)
    k_gemm_packed<N_FEAT, HIDDEN><<<blk((long long)m_tiles * 32, TPB), TPB, 0, stream>>>(
        xb, Wb1, h1, m_tiles);
    k_self_init<<<blk((long long)N_NODES * HIDDEN, TPB), TPB, 0, stream>>>(
        h1, dinv, agg1, N_NODES * HIDDEN, HIDDEN);
    k_agg_edges<HIDDEN><<<blk((long long)N_EDGES * 32, TPB), TPB, 0, stream>>>(
        h1, rows, cols, norm, agg1, N_EDGES);
    k_pack_relu<<<blk((long long)N_NODES * HIDDEN, TPB), TPB, 0, stream>>>(
        agg1, b1, h2b, N_NODES * HIDDEN);

    // ---- 4) layer 2: g2 = relu(h) @ W2 (WMMA), then aggregate into d_out
    k_gemm_packed<HIDDEN, N_CLASSES><<<blk((long long)m_tiles * 32, TPB), TPB, 0, stream>>>(
        h2b, Wb2, g2, m_tiles);
    k_out_init<<<blk((long long)N_NODES * N_CLASSES, TPB), TPB, 0, stream>>>(
        g2, dinv, b2, out, N_NODES * N_CLASSES, N_CLASSES);
    k_agg_edges<N_CLASSES><<<blk((long long)N_EDGES * 32, TPB), TPB, 0, stream>>>(
        g2, rows, cols, norm, out, N_EDGES);
}
